// Diffusion_16381005267254
// MI455X (gfx1250) — matrix operand order repaired and checked by reading, not compiled, not run on hardware
//
#include <hip/hip_runtime.h>

// Problem dims (match reference)
#define N_NODES 2048
#define N_CH    512
// expm-action parameters: y <- [p_m(-tL/s)]^s x
#define S_SEG   16
#define M_TERMS 16

typedef __attribute__((ext_vector_type(16))) __bf16 v16bf;
typedef __attribute__((ext_vector_type(8)))  float  v8f;
typedef __attribute__((ext_vector_type(4)))  unsigned int v4u;
typedef __attribute__((ext_vector_type(8)))  unsigned int v8u;

union FragU { uint4 q[2]; v16bf v; };

// float -> bf16 round-to-nearest-even (bit trick), bf16 -> float
__device__ __forceinline__ unsigned short f2bf(float f) {
  unsigned int u = __float_as_uint(f);
  u += 0x7FFFu + ((u >> 16) & 1u);
  return (unsigned short)(u >> 16);
}
__device__ __forceinline__ float bf2f(unsigned short h) {
  return __uint_as_float(((unsigned int)h) << 16);
}

// ---- Tensor Data Mover: 2D tile load Global->LDS (CDNA5 ISA).
// D# group1 (256b), constant across the K loop:
//   data_size=1 (2B elements), pad_enable=1, pad_interval=3 (16 DWORDs = 64B,
//   one 32-bf16 row), pad_amount=3 (4 DWORDs = 16B = 8 bf16) -> LDS row pitch
//   40 bf16.  tensor_dim0=tile_dim0=32 (K, contiguous), tensor_dim1=tile_dim1=64
//   (M rows), tensor_dim0_stride=2048 (true row pitch of L in elements).
__device__ __forceinline__ v8u tdm_group1_Atile() {
  v8u g;
  const unsigned TD0 = 32u, TD1 = 64u, TILE0 = 32u, TILE1 = 64u;
  const unsigned long long STR0 = 2048ull;
  g.s0 = (1u << 16) | (1u << 20) | (3u << 22) | (3u << 25); // data_size|pad cfg
  g.s1 = (TD0 & 0xFFFFu) << 16;                             // tensor_dim0[15:0] @ bit48
  g.s2 = ((TD0 >> 16) & 0xFFFFu) | ((TD1 & 0xFFFFu) << 16); // dim0 hi | dim1 lo
  g.s3 = ((TD1 >> 16) & 0xFFFFu) | (TILE0 << 16);           // dim1 hi | tile_dim0
  g.s4 = TILE1;                                             // tile_dim1 (tile_dim2=0)
  g.s5 = (unsigned)(STR0 & 0xFFFFFFFFull);                  // dim0_stride lo
  g.s6 = (unsigned)((STR0 >> 32) & 0xFFFFull);              // dim0_stride hi (dim1_stride=0)
  g.s7 = 0u;
  return g;
}

// D# group0: count=1, lds_addr, 57-bit global tile address, type=2 ("image").
__device__ __forceinline__ void tdm_load_tile(unsigned lds_addr, const void* gptr,
                                              v8u g1) {
  unsigned long long ga = (unsigned long long)(uintptr_t)gptr;
  v4u g0;
  g0.x = 1u;                                             // count=1
  g0.y = lds_addr;                                       // lds_addr [63:32]
  g0.z = (unsigned)(ga & 0xFFFFFFFFull);                 // global_addr [95:64]
  g0.w = (unsigned)((ga >> 32) & 0x01FFFFFFull)          // global_addr [120:96]
         | 0x80000000u;                                  // type=2 [127:126]
  asm volatile("tensor_load_to_lds %0, %1" :: "s"(g0), "s"(g1) : "memory");
}

// Split a float array into bf16 hi/lo pair (lo = x - float(hi)), grid-stride.
__global__ void split_f32_to_bf16x2(const float* __restrict__ src,
                                    unsigned short* __restrict__ hi,
                                    unsigned short* __restrict__ lo,
                                    size_t n) {
  size_t stride = (size_t)gridDim.x * blockDim.x;
  for (size_t i = (size_t)blockIdx.x * blockDim.x + threadIdx.x; i < n; i += stride) {
    float a = src[i];
    unsigned short h = f2bf(a);
    hi[i] = h;
    lo[i] = f2bf(a - bf2f(h));
  }
}

// y = x, and split x into the first Taylor-term buffer (k=0 term).
__global__ void init_state(const float* __restrict__ x,
                           float* __restrict__ y,
                           unsigned short* __restrict__ hi,
                           unsigned short* __restrict__ lo,
                           size_t n) {
  size_t stride = (size_t)gridDim.x * blockDim.x;
  for (size_t i = (size_t)blockIdx.x * blockDim.x + threadIdx.x; i < n; i += stride) {
    float a = x[i];
    y[i] = a;
    unsigned short h = f2bf(a);
    hi[i] = h;
    lo[i] = f2bf(a - bf2f(h));
  }
}

// One Taylor step: W = scale * (L @ V), y += W, and emit W's bf16 hi/lo split
// for the next step.  scale = -max(t,1e-8)/(S_SEG*k), computed on device.
// Split-bf16 GEMM: L@V ~= Lhi@Vhi + Lhi@Vlo + Llo@Vhi, f32 accumulate.
// C tile 64x64 per 256-thread block; 8 waves, each wave a 16(M)x32(N) strip.
// A tiles staged via TDM (tensor_load_to_lds), B tiles via VMEM + LDS transpose.
__global__ __launch_bounds__(256)
void heat_apply_gemm(const unsigned short* __restrict__ Ahi,   // L_hi [2048x2048] bf16
                     const unsigned short* __restrict__ Alo,   // L_lo
                     const unsigned short* __restrict__ Bhi,   // v_hi [2048x512] bf16
                     const unsigned short* __restrict__ Blo,   // v_lo
                     float* __restrict__ y,                    // accumulator [2048x512]
                     unsigned short* __restrict__ Whi,         // next-term hi
                     unsigned short* __restrict__ Wlo,         // next-term lo
                     const float* __restrict__ tptr,
                     float inv_sk) {
  // LDS tiles, rows padded to 40 bf16 (80B): TDM pads A, we pad B manually.
  __shared__ unsigned short sAh[64 * 40];
  __shared__ unsigned short sAl[64 * 40];
  __shared__ unsigned short sBh[64 * 40];   // stored transposed: [n][k]
  __shared__ unsigned short sBl[64 * 40];

  const int tid   = threadIdx.x;
  const int lane  = tid & 31;
  const int wave  = tid >> 5;
  const int m_off = (wave >> 1) << 4;   // 0,16,32,48
  const int n_off = (wave & 1) << 5;    // 0,32
  const int m_blk = blockIdx.y << 6;
  const int n_blk = blockIdx.x << 6;

  // cooperative global->LDS indices for B
  const int b_row = tid >> 3;           // 0..31   (K)
  const int b_col = (tid & 7) << 3;     // 0..56   (N)

  const int mlane = lane & 15;
  const int khalf = lane >> 4;

  const v8u g1 = tdm_group1_Atile();
  const unsigned ldsAh = (unsigned)(uintptr_t)&sAh[0];
  const unsigned ldsAl = (unsigned)(uintptr_t)&sAl[0];

  v8f c0 = {};  // N sub-tile 0
  v8f c1 = {};  // N sub-tile 16

  for (int kt = 0; kt < N_NODES; kt += 32) {
    __syncthreads();
    // ---- A tiles (64 x 32) hi & lo: Tensor Data Mover, wave 0 issues DMA
    if (wave == 0) {
      tdm_load_tile(ldsAh, Ahi + (size_t)m_blk * N_NODES + kt, g1);
      tdm_load_tile(ldsAl, Alo + (size_t)m_blk * N_NODES + kt, g1);
    }
    // ---- B tile (32K x 64N), hi & lo, stored transposed [n][k]
    {
      size_t g = (size_t)(kt + b_row) * N_CH + (size_t)(n_blk + b_col);
      uint4 h = *(const uint4*)(Bhi + g);
      uint4 l = *(const uint4*)(Blo + g);
      const unsigned short* hp = (const unsigned short*)&h;
      const unsigned short* lp = (const unsigned short*)&l;
#pragma unroll
      for (int i = 0; i < 8; ++i) {
        sBh[(b_col + i) * 40 + b_row] = hp[i];
        sBl[(b_col + i) * 40 + b_row] = lp[i];
      }
    }
    if (kt + 32 < N_NODES) {  // gfx1250 global_prefetch_b8 for next B K-tile
      __builtin_prefetch(Bhi + (size_t)(kt + 32 + b_row) * N_CH + (n_blk + b_col), 0, 3);
    }
    if (wave == 0) __builtin_amdgcn_s_wait_tensorcnt(0);  // DMA landed in LDS
    __syncthreads();

    // ---- fragments (wave32 ISA layouts)
    // A 16x32 bf16: lane m=lane&15, kh=lane>>4; elems e<8 -> K=8*kh+e,
    // e>=8 -> K=16+8*kh+(e-8): two contiguous 16B chunks per lane.
    FragU ah, al, bh0, bl0, bh1, bl1;
    const int ar = (m_off + mlane) * 40;
    ah.q[0] = *(const uint4*)&sAh[ar + 8 * khalf];
    ah.q[1] = *(const uint4*)&sAh[ar + 16 + 8 * khalf];
    al.q[0] = *(const uint4*)&sAl[ar + 8 * khalf];
    al.q[1] = *(const uint4*)&sAl[ar + 16 + 8 * khalf];
    // B 32x16 bf16: lane n=lane&15, kh=lane>>4; elems e<8 -> K=8*kh+e,
    // e>=8 -> K=16+8*kh+(e-8): two 16B chunks per lane in the transposed LDS tile.
    const int br0 = (n_off + mlane) * 40 + 8 * khalf;
    const int br1 = (n_off + 16 + mlane) * 40 + 8 * khalf;
    bh0.q[0] = *(const uint4*)&sBh[br0]; bh0.q[1] = *(const uint4*)&sBh[br0 + 16];
    bl0.q[0] = *(const uint4*)&sBl[br0]; bl0.q[1] = *(const uint4*)&sBl[br0 + 16];
    bh1.q[0] = *(const uint4*)&sBh[br1]; bh1.q[1] = *(const uint4*)&sBh[br1 + 16];
    bl1.q[0] = *(const uint4*)&sBl[br1]; bl1.q[1] = *(const uint4*)&sBl[br1 + 16];

    // split-bf16 product: hi*hi + hi*lo + lo*hi, f32 accumulate
    c0 = __builtin_amdgcn_wmma_f32_16x16x32_bf16(false, ah.v, false, bh0.v, (short)0, c0, false, false);
    c0 = __builtin_amdgcn_wmma_f32_16x16x32_bf16(false, ah.v, false, bl0.v, (short)0, c0, false, false);
    c0 = __builtin_amdgcn_wmma_f32_16x16x32_bf16(false, al.v, false, bh0.v, (short)0, c0, false, false);
    c1 = __builtin_amdgcn_wmma_f32_16x16x32_bf16(false, ah.v, false, bh1.v, (short)0, c1, false, false);
    c1 = __builtin_amdgcn_wmma_f32_16x16x32_bf16(false, ah.v, false, bl1.v, (short)0, c1, false, false);
    c1 = __builtin_amdgcn_wmma_f32_16x16x32_bf16(false, al.v, false, bh1.v, (short)0, c1, false, false);
  }

  // ---- fused epilogue: scale, y += term, re-split term to bf16 hi/lo
  const float tt    = fmaxf(tptr[0], 1e-8f);
  const float scale = -tt * inv_sk;

#pragma unroll
  for (int r = 0; r < 8; ++r) {
    const int row = m_blk + m_off + r + 8 * khalf;   // C/D layout: VGPR r -> M=r / r+8
    {
      const int col  = n_blk + n_off + mlane;
      const size_t o = (size_t)row * N_CH + col;
      const float v  = scale * c0[r];
      y[o] += v;
      unsigned short h = f2bf(v);
      Whi[o] = h;
      Wlo[o] = f2bf(v - bf2f(h));
    }
    {
      const int col  = n_blk + n_off + 16 + mlane;
      const size_t o = (size_t)row * N_CH + col;
      const float v  = scale * c1[r];
      y[o] += v;
      unsigned short h = f2bf(v);
      Whi[o] = h;
      Wlo[o] = f2bf(v - bf2f(h));
    }
  }
}

extern "C" void kernel_launch(void* const* d_in, const int* in_sizes, int n_in,
                              void* d_out, int out_size, void* d_ws, size_t ws_size,
                              hipStream_t stream) {
  const float* x = (const float*)d_in[0];   // [2048, 512]
  const float* L = (const float*)d_in[1];   // [2048, 2048]
  const float* t = (const float*)d_in[2];   // scalar
  float* y = (float*)d_out;                 // [2048, 512]

  const size_t NN = (size_t)N_NODES * N_NODES;
  const size_t ND = (size_t)N_NODES * N_CH;

  // workspace layout (24 MB): L split (16 MB) + two term buffers (2x4 MB)
  char* ws = (char*)d_ws;
  unsigned short* Lhi   = (unsigned short*)(ws);
  unsigned short* Llo   = (unsigned short*)(ws + NN * 2);
  unsigned short* va_hi = (unsigned short*)(ws + NN * 4);
  unsigned short* va_lo = (unsigned short*)(ws + NN * 4 + ND * 2);
  unsigned short* vb_hi = (unsigned short*)(ws + NN * 4 + ND * 4);
  unsigned short* vb_lo = (unsigned short*)(ws + NN * 4 + ND * 6);

  split_f32_to_bf16x2<<<4096, 256, 0, stream>>>(L, Lhi, Llo, NN);
  init_state<<<2048, 256, 0, stream>>>(x, y, va_hi, va_lo, ND);

  dim3 grid(N_CH / 64, N_NODES / 64);   // 8 x 32 = 256 blocks
  dim3 block(256);

  unsigned short* cur_hi = va_hi; unsigned short* cur_lo = va_lo;
  unsigned short* nxt_hi = vb_hi; unsigned short* nxt_lo = vb_lo;

  for (int seg = 0; seg < S_SEG; ++seg) {
    if (seg > 0) {  // start next segment's Taylor series from current y
      split_f32_to_bf16x2<<<2048, 256, 0, stream>>>(y, cur_hi, cur_lo, ND);
    }
    for (int k = 1; k <= M_TERMS; ++k) {
      float inv_sk = 1.0f / (float)(S_SEG * k);
      heat_apply_gemm<<<grid, block, 0, stream>>>(Lhi, Llo, cur_hi, cur_lo,
                                                  y, nxt_hi, nxt_lo, t, inv_sk);
      unsigned short* th = cur_hi; cur_hi = nxt_hi; nxt_hi = th;
      unsigned short* tl = cur_lo; cur_lo = nxt_lo; nxt_lo = tl;
    }
  }
}
